// FAGCNNet_28991029248703
// MI455X (gfx1250) — matrix operand order repaired and checked by reading, not compiled, not run on hardware
//
#include <hip/hip_runtime.h>
#include <hip/hip_bf16.h>

#define NNODES 100000
#define EDGES  1600000
#define FIN    500
#define DIM    64
#define NCLS   40
#define EPS_F  0.3f

typedef __attribute__((ext_vector_type(16))) _Float16 v16h;
typedef __attribute__((ext_vector_type(8)))  _Float16 v8h;
typedef __attribute__((ext_vector_type(8)))  float    v8f;

// ---------------------------------------------------------------------------
// Edge weight: sigmoid with |w|>0 mask
// ---------------------------------------------------------------------------
__global__ __launch_bounds__(256) void ew_kernel(const float* __restrict__ w,
                                                 float* __restrict__ ew) {
    int i = blockIdx.x * 256 + threadIdx.x;
    if (i < EDGES) {
        float v = w[i];
        ew[i] = (fabsf(v) > 0.0f) ? 1.0f / (1.0f + expf(-v)) : 0.0f;
    }
}

// ---------------------------------------------------------------------------
// GEMM1: h = relu(x @ t1_w^T + b)   [100000 x 500] x [500 x 64]
// f16 WMMA, f32 accumulate. Block = 256 threads (8 waves), 128 rows per block.
// Each wave owns a 16-row M tile and all 4 N tiles (64 cols).
// ---------------------------------------------------------------------------
#define ASTR 40   // LDS row stride in halfs (breaks bank conflicts, 16B aligned)
#define BSTR 40

__global__ __launch_bounds__(256) void gemm1_kernel(const float* __restrict__ x,
                                                    const float* __restrict__ w,
                                                    const float* __restrict__ bias,
                                                    float* __restrict__ h) {
    __shared__ __align__(16) _Float16 As[128 * ASTR];  // 128 x 32 tile of x (f16)
    __shared__ __align__(16) _Float16 Bs[64 * BSTR];   // 64  x 32 tile of w^T (f16)

    const int tid  = threadIdx.x;
    const int lane = tid & 31;
    const int wave = tid >> 5;
    const int blockRow = blockIdx.x * 128;

    v8f zero = {0.f, 0.f, 0.f, 0.f, 0.f, 0.f, 0.f, 0.f};
    v8f acc[4];
    #pragma unroll
    for (int nt = 0; nt < 4; ++nt) acc[nt] = zero;

    const int nChunks = (FIN + 31) / 32;  // 16 (last chunk covers K=480..499, zero padded)

    for (int kc = 0; kc < nChunks; ++kc) {
        const int k0 = kc * 32;

        // ---- stage A tile: 128 rows x 32 K, fp32 -> f16 ----
        {
            const int row  = tid >> 1;          // 0..127
            const int ch   = (tid & 1) * 16;    // 0 or 16
            const int grow = blockRow + row;
            float v[16];
            if (grow < NNODES && (k0 + 32) <= FIN) {
                const float* p = x + (size_t)grow * FIN + k0 + ch;
                #pragma unroll
                for (int j = 0; j < 16; ++j) v[j] = p[j];
                // prefetch next chunk's segment of this row
                if (k0 + 64 <= FIN)
                    __builtin_prefetch(p + 32, 0, 1);
            } else {
                #pragma unroll
                for (int j = 0; j < 16; ++j) {
                    int gk = k0 + ch + j;
                    v[j] = (grow < NNODES && gk < FIN) ? x[(size_t)grow * FIN + gk] : 0.0f;
                }
            }
            #pragma unroll
            for (int j = 0; j < 16; ++j)
                As[row * ASTR + ch + j] = (_Float16)v[j];
        }
        // ---- stage B tile: 64 cols x 32 K from t1_w [64 x 500] ----
        {
            const int n  = tid >> 2;            // 0..63
            const int ch = (tid & 3) * 8;       // 0,8,16,24
            #pragma unroll
            for (int j = 0; j < 8; ++j) {
                int gk = k0 + ch + j;
                float v = (gk < FIN) ? w[n * FIN + gk] : 0.0f;
                Bs[n * BSTR + ch + j] = (_Float16)v;
            }
        }
        __syncthreads();

        // ---- A fragment: lane r<16 -> row r, K = {0..7,16..23}; hi lanes K+8 ----
        const int r    = lane & 15;
        const int kbA  = (lane >> 4) * 8;
        const int arow = wave * 16 + r;
        v8h alo = *(const v8h*)&As[arow * ASTR + kbA];
        v8h ahi = *(const v8h*)&As[arow * ASTR + kbA + 16];
        v16h a = __builtin_shufflevector(alo, ahi,
                  0, 1, 2, 3, 4, 5, 6, 7, 8, 9, 10, 11, 12, 13, 14, 15);

        // ---- B fragments + WMMA per N tile ----
        const int kbB = (lane >> 4) * 8;        // lanes 0-15: K {0..7,16..23}; 16-31: K {8..15,24..31}
        #pragma unroll
        for (int nt = 0; nt < 4; ++nt) {
            const int bn = nt * 16 + r;
            v8h blo = *(const v8h*)&Bs[bn * BSTR + kbB];
            v8h bhi = *(const v8h*)&Bs[bn * BSTR + kbB + 16];
            v16h b = __builtin_shufflevector(blo, bhi,
                      0, 1, 2, 3, 4, 5, 6, 7, 8, 9, 10, 11, 12, 13, 14, 15);
            acc[nt] = __builtin_amdgcn_wmma_f32_16x16x32_f16(
                false, a, false, b, (short)0, acc[nt], false, false);
        }
        __syncthreads();
    }

    // ---- store D with bias + relu. lane -> N = nt*16+(lane&15); M = r (+8 hi) ----
    const int r15 = lane & 15;
    const int mhi = (lane >> 4) * 8;
    #pragma unroll
    for (int nt = 0; nt < 4; ++nt) {
        const int n  = nt * 16 + r15;
        const float bv = bias[n];
        #pragma unroll
        for (int rr = 0; rr < 8; ++rr) {
            const int grow = blockRow + wave * 16 + rr + mhi;
            if (grow < NNODES) {
                float v = acc[nt][rr] + bv;
                h[(size_t)grow * DIM + n] = v > 0.0f ? v : 0.0f;
            }
        }
    }
}

// ---------------------------------------------------------------------------
// Per-node dots: hl = h.att_l, hr = h.att_r. One wave32 per node.
// ---------------------------------------------------------------------------
__global__ __launch_bounds__(256) void dots_kernel(const float* __restrict__ h,
                                                   const float* __restrict__ attl,
                                                   const float* __restrict__ attr,
                                                   float* __restrict__ hl,
                                                   float* __restrict__ hr) {
    int gid  = blockIdx.x * 256 + threadIdx.x;
    int node = gid >> 5;
    int lane = gid & 31;
    if (node >= NNODES) return;
    float h0 = h[(size_t)node * DIM + lane];
    float h1 = h[(size_t)node * DIM + lane + 32];
    float pl = h0 * attl[lane] + h1 * attl[lane + 32];
    float pr = h0 * attr[lane] + h1 * attr[lane + 32];
    #pragma unroll
    for (int off = 16; off > 0; off >>= 1) {
        pl += __shfl_down(pl, off, 32);
        pr += __shfl_down(pr, off, 32);
    }
    if (lane == 0) { hl[node] = pl; hr[node] = pr; }
}

// ---------------------------------------------------------------------------
// Init accumulator: hnext = EPS * raw
// ---------------------------------------------------------------------------
__global__ __launch_bounds__(256) void scale_kernel(const float* __restrict__ raw,
                                                    float* __restrict__ out) {
    long i = (long)blockIdx.x * 256 + threadIdx.x;
    if (i < (long)NNODES * DIM) out[i] = EPS_F * raw[i];
}

// ---------------------------------------------------------------------------
// Edge scatter: one wave32 per edge, lane d handles dims d and d+32.
// hnext[dst] += tanh(hl[src]+hr[dst]) * ew * hcur[src]   (L2-resident atomics)
// ---------------------------------------------------------------------------
__global__ __launch_bounds__(256) void edge_kernel(const int* __restrict__ src,
                                                   const int* __restrict__ dst,
                                                   const float* __restrict__ ew,
                                                   const float* __restrict__ hl,
                                                   const float* __restrict__ hr,
                                                   const float* __restrict__ hcur,
                                                   float* __restrict__ hnext) {
    int gid  = blockIdx.x * 256 + threadIdx.x;
    int e    = gid >> 5;
    int lane = gid & 31;
    if (e >= EDGES) return;
    float w = ew[e];
    if (w == 0.0f) return;
    int s = src[e];
    int d = dst[e];
    float c = tanhf(hl[s] + hr[d]) * w;
    size_t sb = (size_t)s * DIM;
    size_t db = (size_t)d * DIM;
    atomicAdd(&hnext[db + lane],      c * hcur[sb + lane]);
    atomicAdd(&hnext[db + lane + 32], c * hcur[sb + lane + 32]);
}

// ---------------------------------------------------------------------------
// Classifier + log_softmax: thread per node, t2_w staged in LDS.
// ---------------------------------------------------------------------------
__global__ __launch_bounds__(256) void out_kernel(const float* __restrict__ h,
                                                  const float* __restrict__ w2,
                                                  const float* __restrict__ b2,
                                                  float* __restrict__ out) {
    __shared__ float Wl[NCLS * DIM];
    __shared__ float Bl[NCLS];
    for (int i = threadIdx.x; i < NCLS * DIM; i += 256) Wl[i] = w2[i];
    if (threadIdx.x < NCLS) Bl[threadIdx.x] = b2[threadIdx.x];
    __syncthreads();

    int n = blockIdx.x * 256 + threadIdx.x;
    if (n >= NNODES) return;

    float logits[NCLS];
    #pragma unroll
    for (int c = 0; c < NCLS; ++c) logits[c] = Bl[c];
    for (int d = 0; d < DIM; ++d) {
        float hd = h[(size_t)n * DIM + d];
        #pragma unroll
        for (int c = 0; c < NCLS; ++c) logits[c] += hd * Wl[c * DIM + d];
    }
    float m = logits[0];
    #pragma unroll
    for (int c = 1; c < NCLS; ++c) m = fmaxf(m, logits[c]);
    float s = 0.0f;
    #pragma unroll
    for (int c = 0; c < NCLS; ++c) s += expf(logits[c] - m);
    float lse = logf(s);
    #pragma unroll
    for (int c = 0; c < NCLS; ++c) out[(size_t)n * NCLS + c] = logits[c] - m - lse;
}

// ---------------------------------------------------------------------------
extern "C" void kernel_launch(void* const* d_in, const int* in_sizes, int n_in,
                              void* d_out, int out_size, void* d_ws, size_t ws_size,
                              hipStream_t stream) {
    const float* x    = (const float*)d_in[0];
    const int*   ei   = (const int*)d_in[1];
    const float* ewt  = (const float*)d_in[2];
    const float* t1w  = (const float*)d_in[3];
    const float* t1b  = (const float*)d_in[4];
    const float* al1  = (const float*)d_in[5];
    const float* ar1  = (const float*)d_in[6];
    const float* al2  = (const float*)d_in[7];
    const float* ar2  = (const float*)d_in[8];
    const float* t2w  = (const float*)d_in[9];
    const float* t2b  = (const float*)d_in[10];
    float* out = (float*)d_out;

    // workspace layout (fp32)
    float* raw = (float*)d_ws;                       // N*D  raw = relu(x@W1^T+b1)
    float* h1  = raw + (size_t)NNODES * DIM;         // N*D  layer-1 output
    float* h2  = h1  + (size_t)NNODES * DIM;         // N*D  layer-2 output
    float* hl  = h2  + (size_t)NNODES * DIM;         // N
    float* hr  = hl  + NNODES;                       // N
    float* ew  = hr  + NNODES;                       // E
    const int* src = ei;
    const int* dst = ei + EDGES;

    const long nd = (long)NNODES * DIM;

    ew_kernel<<<(EDGES + 255) / 256, 256, 0, stream>>>(ewt, ew);
    gemm1_kernel<<<(NNODES + 127) / 128, 256, 0, stream>>>(x, t1w, t1b, raw);

    // FAConv layer 1: raw -> h1
    dots_kernel<<<((long)NNODES * 32 + 255) / 256, 256, 0, stream>>>(raw, al1, ar1, hl, hr);
    scale_kernel<<<(nd + 255) / 256, 256, 0, stream>>>(raw, h1);
    edge_kernel<<<((long)EDGES * 32 + 255) / 256, 256, 0, stream>>>(src, dst, ew, hl, hr, raw, h1);

    // FAConv layer 2: h1 -> h2  (raw term unchanged)
    dots_kernel<<<((long)NNODES * 32 + 255) / 256, 256, 0, stream>>>(h1, al2, ar2, hl, hr);
    scale_kernel<<<(nd + 255) / 256, 256, 0, stream>>>(raw, h2);
    edge_kernel<<<((long)EDGES * 32 + 255) / 256, 256, 0, stream>>>(src, dst, ew, hl, hr, h1, h2);

    out_kernel<<<(NNODES + 255) / 256, 256, 0, stream>>>(h2, t2w, t2b, out);
}
